// BlockDiagonalLowRankLinear_63402307224415
// MI455X (gfx1250) — matrix operand order repaired and checked by reading, not compiled, not run on hardware
//
#include <hip/hip_runtime.h>
#include <hip/hip_bf16.h>

// ---------------------------------------------------------------------------
// BlockDiagonalLowRankLinear for MI455X (gfx1250, wave32, WMMA)
// y = (BlockDiag(W) x + U V^T x) * alpha + bias
// x:[8192,4096] blocks:[16,256,256] U:[4096,64] V:[4096,64] bias:[4096]
// Strategy: bf16 hi/lo split GEMMs on v_wmma_f32_16x16x32_bf16 (3 products),
// weights pre-split once into workspace, x split on the fly with a cheap
// truncation split (v_and + v_sub + v_perm packing). Pass 1 software-pipelines
// the x loads across K iterations.
// ---------------------------------------------------------------------------

typedef __attribute__((ext_vector_type(16))) __bf16 v16bf;
typedef __attribute__((ext_vector_type(8)))  float  v8f;

namespace {

constexpr int D_IN  = 4096;
constexpr int D_OUT = 4096;
constexpr int NB    = 16;
constexpr int BI    = 256;
constexpr int BO    = 256;
constexpr int RNK   = 64;
constexpr int NROWS = 8192;   // 4 * 2048

union Frag {
  v16bf v;
  unsigned short s[16];
  unsigned u[8];
  uint4 q[2];
};

__device__ __forceinline__ unsigned short f32_to_bf16(float f) {
  unsigned u = __float_as_uint(f);
  unsigned r = u + 0x7FFFu + ((u >> 16) & 1u);   // round-to-nearest-even
  return (unsigned short)(r >> 16);
}
__device__ __forceinline__ float bf16_to_f32(unsigned short s) {
  return __uint_as_float(((unsigned)s) << 16);
}
__device__ __forceinline__ void split2(float f, unsigned short& h, unsigned short& l) {
  h = f32_to_bf16(f);
  l = f32_to_bf16(f - bf16_to_f32(h));
}

__device__ __forceinline__ v8f vzero8() {
  v8f z;
#pragma unroll
  for (int i = 0; i < 8; ++i) z[i] = 0.0f;
  return z;
}

__device__ __forceinline__ v8f wmma_bf16(v16bf a, v16bf b, v8f c) {
  // D = A(16x32 bf16) * B(32x16 bf16) + C(16x16 f32)
  return __builtin_amdgcn_wmma_f32_16x16x32_bf16(
      /*neg_a=*/false, a, /*neg_b=*/false, b,
      /*c_mod=*/(short)0, c, /*reuse_a=*/false, /*reuse_b=*/false);
}

// ---- A fragment handling (f32 source, truncation hi/lo bf16 split) --------
// A layout: lane holds row = lane&15; halfwords 0..7 -> K = kb..kb+7,
// halfwords 8..15 -> K = kb+16..kb+23, kb = k0 + (lane>>4)*8.

// Stage 1: pure loads (16 f32 per lane as 4x float4).
__device__ __forceinline__ void load_a_raw(const float* __restrict__ base, int ld,
                                           int row0, int k0, float4 r[4]) {
  const int lane = threadIdx.x & 31;
  const float* p = base + (size_t)(row0 + (lane & 15)) * (size_t)ld
                        + (size_t)(k0 + ((lane >> 4) << 3));
  r[0] = *(const float4*)(p + 0);
  r[1] = *(const float4*)(p + 4);
  r[2] = *(const float4*)(p + 16);
  r[3] = *(const float4*)(p + 20);
}

// Stage 2: split + pack. h = f & 0xFFFF0000 (exact residual l = f - h),
// l truncated to bf16; two top-halfwords packed per dword via one v_perm_b32.
__device__ __forceinline__ void convert_a(const float4 r[4], v16bf& fh, v16bf& fl) {
  float vv[16] = { r[0].x, r[0].y, r[0].z, r[0].w, r[1].x, r[1].y, r[1].z, r[1].w,
                   r[2].x, r[2].y, r[2].z, r[2].w, r[3].x, r[3].y, r[3].z, r[3].w };
  unsigned hu[16];
  unsigned lu[16];
#pragma unroll
  for (int i = 0; i < 16; ++i) {
    unsigned ui = __float_as_uint(vv[i]);
    unsigned hb = ui & 0xFFFF0000u;                        // truncated bf16 hi
    hu[i] = hb;
    lu[i] = __float_as_uint(vv[i] - __uint_as_float(hb));  // exact residual
  }
  Frag H, L;
#pragma unroll
  for (int j = 0; j < 8; ++j) {
    // dword j = { elem(2j+1).top16 , elem(2j).top16 }
    H.u[j] = __builtin_amdgcn_perm(hu[2 * j + 1], hu[2 * j], 0x07060302u);
    L.u[j] = __builtin_amdgcn_perm(lu[2 * j + 1], lu[2 * j], 0x07060302u);
  }
  fh = H.v;
  fl = L.v;
}

__device__ __forceinline__ void load_a_f32(const float* __restrict__ base, int ld,
                                           int row0, int k0, v16bf& fh, v16bf& fl) {
  float4 r[4];
  load_a_raw(base, ld, row0, k0, r);
  convert_a(r, fh, fl);
}

// A fragment from pre-split bf16 storage (rows of length ld).
__device__ __forceinline__ v16bf load_a_u16(const unsigned short* __restrict__ base,
                                            int ld, int row0, int k0) {
  const int lane = threadIdx.x & 31;
  const unsigned short* p = base + (size_t)(row0 + (lane & 15)) * (size_t)ld
                                 + (size_t)(k0 + ((lane >> 4) << 3));
  Frag u;
  u.q[0] = *(const uint4*)(p);        // K kb..kb+7
  u.q[1] = *(const uint4*)(p + 16);   // K kb+16..kb+23
  return u.v;
}

// B fragment from pre-split bf16 storage.
// B layout: lane holds col = lane&15; halfwords 0..7 -> K = kb..kb+7,
// halfwords 8..15 -> K = kb+16..kb+23, kb = k0 + (lane>>4)*8.
__device__ __forceinline__ v16bf load_b_u16(const unsigned short* __restrict__ base,
                                            int ld, int col0, int k0) {
  const int lane = threadIdx.x & 31;
  const unsigned short* p = base + (size_t)(col0 + (lane & 15)) * (size_t)ld
                                 + (size_t)(k0 + ((lane >> 4) << 3));
  Frag u;
  u.q[0] = *(const uint4*)(p);
  u.q[1] = *(const uint4*)(p + 16);
  return u.v;
}

} // namespace

// ---------------------------------------------------------------------------
// Prep kernels
// ---------------------------------------------------------------------------
__global__ void prep_split(const float* __restrict__ in,
                           unsigned short* __restrict__ h,
                           unsigned short* __restrict__ l, int n) {
  int i = blockIdx.x * blockDim.x + threadIdx.x;
  if (i < n) {
    unsigned short hh, ll;
    split2(in[i], hh, ll);
    h[i] = hh;
    l[i] = ll;
  }
}

// V:[D_IN][RNK] -> Vt:[RNK][D_IN] split into bf16 hi/lo
__global__ void prep_transpose_split_V(const float* __restrict__ V,
                                       unsigned short* __restrict__ Vth,
                                       unsigned short* __restrict__ Vtl) {
  int t = blockIdx.x * blockDim.x + threadIdx.x;
  if (t < D_IN * RNK) {
    int r = t & (RNK - 1);
    int i = t >> 6;
    unsigned short h, l;
    split2(V[t], h, l);
    Vth[(size_t)r * D_IN + i] = h;
    Vtl[(size_t)r * D_IN + i] = l;
  }
}

// ---------------------------------------------------------------------------
// Pass 1: T = x @ V  -> stored pre-split bf16 hi/lo, T:[NROWS][RNK]
// 1 wave computes 16 rows x 64 cols; blockDim=128 (4 waves) -> 64 rows / WG.
// Software-pipelined: raw x loads for iteration k+1 issue during iteration k.
// ---------------------------------------------------------------------------
__global__ void __launch_bounds__(128)
pass1_xV(const float* __restrict__ x,
         const unsigned short* __restrict__ Vth, const unsigned short* __restrict__ Vtl,
         unsigned short* __restrict__ Th, unsigned short* __restrict__ Tl) {
  const int wave = threadIdx.x >> 5;
  const int lane = threadIdx.x & 31;
  const int rowBase = (blockIdx.x * 4 + wave) * 16;

  v8f acc[4];
#pragma unroll
  for (int ni = 0; ni < 4; ++ni) acc[ni] = vzero8();

  float4 ra[4];
  load_a_raw(x, D_IN, rowBase, 0, ra);   // prime the pipeline

  for (int k0 = 0; k0 < D_IN; k0 += 32) {
    // Batch all B fragments for this K slab (one clause, one wait point).
    v16bf bh[4], bl[4];
#pragma unroll
    for (int ni = 0; ni < 4; ++ni) {
      bh[ni] = load_b_u16(Vth, D_IN, ni * 16, k0);
      bl[ni] = load_b_u16(Vtl, D_IN, ni * 16, k0);
    }
    // Convert current A while B loads are in flight.
    v16bf ah, al;
    convert_a(ra, ah, al);
    // Prefetch next iteration's raw A (uniform branch; EXEC unchanged).
    if (k0 + 32 < D_IN) load_a_raw(x, D_IN, rowBase, k0 + 32, ra);
#pragma unroll
    for (int ni = 0; ni < 4; ++ni) {
      acc[ni] = wmma_bf16(ah, bh[ni], acc[ni]);
      acc[ni] = wmma_bf16(al, bh[ni], acc[ni]);
      acc[ni] = wmma_bf16(ah, bl[ni], acc[ni]);
    }
  }

  const int hi = lane >> 4, ln = lane & 15;
#pragma unroll
  for (int ni = 0; ni < 4; ++ni) {
#pragma unroll
    for (int r = 0; r < 8; ++r) {
      int row = rowBase + r + hi * 8;
      int col = ni * 16 + ln;
      unsigned short h, l;
      split2(acc[ni][r], h, l);
      Th[(size_t)row * RNK + col] = h;
      Tl[(size_t)row * RNK + col] = l;
    }
  }
}

// ---------------------------------------------------------------------------
// Pass 2: out = (x_blk @ blocks[b]^T + T @ U^T) * alpha + bias
// Grid: x = 16 block-columns, y = 64 row groups of 128 rows.
// 8 waves / WG, each wave owns a 64x64 tile (4x4 WMMA accumulators).
// ---------------------------------------------------------------------------
__global__ void __launch_bounds__(256)
pass2_main(const float* __restrict__ x,
           const unsigned short* __restrict__ Bh, const unsigned short* __restrict__ Bl,
           const unsigned short* __restrict__ Uh, const unsigned short* __restrict__ Ul,
           const unsigned short* __restrict__ Th, const unsigned short* __restrict__ Tl,
           const float* __restrict__ bias, const float* __restrict__ alpha_p,
           float* __restrict__ out) {
  const int wave = threadIdx.x >> 5;
  const int lane = threadIdx.x & 31;
  const int wm = wave >> 2;          // 0..1
  const int wn = wave & 3;           // 0..3
  const int rowBase  = blockIdx.y * 128 + wm * 64;
  const int colBlock = blockIdx.x * BO;        // start of this 256-col block
  const int colBase  = colBlock + wn * 64;     // this wave's 64 cols

  v8f acc[4][4];
#pragma unroll
  for (int mi = 0; mi < 4; ++mi)
#pragma unroll
    for (int ni = 0; ni < 4; ++ni) acc[mi][ni] = vzero8();

  // ---- block-diagonal part: K = 256 over x[:, colBlock..colBlock+255] ----
  const float* xblk = x + colBlock;            // + row*D_IN + k
  for (int k0 = 0; k0 < BI; k0 += 32) {
    v16bf bh[4], bl[4];
#pragma unroll
    for (int ni = 0; ni < 4; ++ni) {
      bh[ni] = load_b_u16(Bh, BI, colBase + ni * 16, k0);  // row o of Bh = global col
      bl[ni] = load_b_u16(Bl, BI, colBase + ni * 16, k0);
    }
#pragma unroll
    for (int mi = 0; mi < 4; ++mi) {
      v16bf ah, al;
      load_a_f32(xblk, D_IN, rowBase + mi * 16, k0, ah, al);
#pragma unroll
      for (int ni = 0; ni < 4; ++ni) {
        acc[mi][ni] = wmma_bf16(ah, bh[ni], acc[mi][ni]);
        acc[mi][ni] = wmma_bf16(al, bh[ni], acc[mi][ni]);
        acc[mi][ni] = wmma_bf16(ah, bl[ni], acc[mi][ni]);
      }
    }
  }

  // ---- low-rank part: K = 64 over T rows, U rows (both pre-split bf16) ----
  for (int k0 = 0; k0 < RNK; k0 += 32) {
    v16bf bh[4], bl[4];
#pragma unroll
    for (int ni = 0; ni < 4; ++ni) {
      bh[ni] = load_b_u16(Uh, RNK, colBase + ni * 16, k0);
      bl[ni] = load_b_u16(Ul, RNK, colBase + ni * 16, k0);
    }
#pragma unroll
    for (int mi = 0; mi < 4; ++mi) {
      v16bf ah = load_a_u16(Th, RNK, rowBase + mi * 16, k0);
      v16bf al = load_a_u16(Tl, RNK, rowBase + mi * 16, k0);
#pragma unroll
      for (int ni = 0; ni < 4; ++ni) {
        acc[mi][ni] = wmma_bf16(ah, bh[ni], acc[mi][ni]);
        acc[mi][ni] = wmma_bf16(al, bh[ni], acc[mi][ni]);
        acc[mi][ni] = wmma_bf16(ah, bl[ni], acc[mi][ni]);
      }
    }
  }

  // ---- epilogue: alpha * acc + bias ----
  const float alpha = alpha_p[0];
  const int hi = lane >> 4, ln = lane & 15;
#pragma unroll
  for (int ni = 0; ni < 4; ++ni) {
    const int col = colBase + ni * 16 + ln;
    const float b = bias[col];
#pragma unroll
    for (int mi = 0; mi < 4; ++mi) {
#pragma unroll
      for (int r = 0; r < 8; ++r) {
        int row = rowBase + mi * 16 + r + hi * 8;
        out[(size_t)row * D_OUT + col] = acc[mi][ni][r] * alpha + b;
      }
    }
  }
}

// ---------------------------------------------------------------------------
// Launch
// ---------------------------------------------------------------------------
extern "C" void kernel_launch(void* const* d_in, const int* in_sizes, int n_in,
                              void* d_out, int out_size, void* d_ws, size_t ws_size,
                              hipStream_t stream) {
  (void)in_sizes; (void)n_in; (void)out_size; (void)ws_size;

  const float* x      = (const float*)d_in[0];
  const float* blocks = (const float*)d_in[1];
  const float* U      = (const float*)d_in[2];
  const float* V      = (const float*)d_in[3];
  const float* bias   = (const float*)d_in[4];
  const float* alpha  = (const float*)d_in[5];
  float* out = (float*)d_out;

  // Workspace layout (8 MB total)
  char* w = (char*)d_ws;
  unsigned short* Bh  = (unsigned short*)(w + 0);                         // 2 MB
  unsigned short* Bl  = (unsigned short*)(w + (size_t)2 * 1024 * 1024);   // 2 MB
  unsigned short* Uh  = (unsigned short*)(w + (size_t)4 * 1024 * 1024);   // 512 KB
  unsigned short* Ul  = (unsigned short*)(w + (size_t)4608 * 1024);       // 512 KB
  unsigned short* Vth = (unsigned short*)(w + (size_t)5 * 1024 * 1024);   // 512 KB
  unsigned short* Vtl = (unsigned short*)(w + (size_t)5632 * 1024);       // 512 KB
  unsigned short* Th  = (unsigned short*)(w + (size_t)6 * 1024 * 1024);   // 1 MB
  unsigned short* Tl  = (unsigned short*)(w + (size_t)7 * 1024 * 1024);   // 1 MB

  // 1) weight prep: split to bf16 hi/lo (and transpose V)
  {
    int n = NB * BO * BI;  // 1,048,576
    prep_split<<<(n + 255) / 256, 256, 0, stream>>>(blocks, Bh, Bl, n);
  }
  {
    int n = D_OUT * RNK;   // 262,144
    prep_split<<<(n + 255) / 256, 256, 0, stream>>>(U, Uh, Ul, n);
  }
  {
    int n = D_IN * RNK;    // 262,144
    prep_transpose_split_V<<<(n + 255) / 256, 256, 0, stream>>>(V, Vth, Vtl);
  }

  // 2) T = x @ V   (128 WGs x 128 threads; 64 rows per WG)
  pass1_xV<<<NROWS / 64, 128, 0, stream>>>(x, Vth, Vtl, Th, Tl);

  // 3) fused blockdiag + lowrank + epilogue
  dim3 grid2(NB, NROWS / 128);
  pass2_main<<<grid2, 256, 0, stream>>>(x, Bh, Bl, Uh, Ul, Th, Tl, bias, alpha, out);
}
